// MambaEncoder_62964220559361
// MI455X (gfx1250) — matrix operand order repaired and checked by reading, not compiled, not run on hardware
//
#include <hip/hip_runtime.h>
#include <hip/hip_bf16.h>

// ---------------- problem constants ----------------
#define BATCH   8
#define SEQ     2048
#define DM      512          // d_model
#define DI      1024         // d_inner
#define DSTATE  64
#define DTR     32           // dt_rank
#define NXPROJ  (DTR + 2*DSTATE)   // 160
#define MTOT    (BATCH*SEQ)        // 16384

typedef __hip_bfloat16 bf16_t;
typedef __attribute__((ext_vector_type(16))) __bf16 v16bf;
typedef __attribute__((ext_vector_type(8)))  float  v8f;

// ---------------- fp32 -> bf16 conversion ----------------
__global__ __launch_bounds__(256) void cvt_f32_bf16(const float* __restrict__ in,
                                                    bf16_t* __restrict__ out, size_t n) {
    size_t i = (size_t)blockIdx.x * 256 + threadIdx.x;
    if (i < n) out[i] = __float2bfloat16(in[i]);
}

// ---------------- generic bf16 WMMA GEMM: C(M,N) = A(M,K) * W(N,K)^T ----------------
// Workgroup = 4 waves (128 threads). Each wave owns a 16(M) x 64(N) strip; the 4
// waves cover a 64x64 output tile. The 64(N) x 32(K) B tile is staged into LDS
// with CDNA5 async global->LDS copies (ASYNCcnt) and double-buffered so the next
// tile's copy overlaps the current tile's WMMAs.
//
// EPI: 0 = store fp32 to Cf
//      1 = store bf16 to Cb0
//      2 = split columns: n < splitN -> Cb0 (bf16), else -> Cb1 (bf16)
//      3 = softplus(c + bias[n]) -> fp32 Cf
template <int EPI>
__global__ __launch_bounds__(128)
void gemm_bf16_wmma(const bf16_t* __restrict__ A, const bf16_t* __restrict__ W,
                    int M, int N, int K,
                    float* __restrict__ Cf,
                    bf16_t* __restrict__ Cb0, bf16_t* __restrict__ Cb1,
                    const float* __restrict__ bias, int splitN)
{
    // Two buffers of: 64 rows x 32 bf16 of K = 4 KB each, held as uint4 (4/row).
    __shared__ uint4 ldsB[2][64 * 4];

    const int wid  = threadIdx.x >> 5;
    const int lane = threadIdx.x & 31;
    const int m0   = blockIdx.x * 64 + wid * 16;
    const int n0   = blockIdx.y * 64;

    // staging role of this thread: 2 threads per W row, 32 bytes each
    const int sn_local = threadIdx.x >> 1;          // 0..63
    const int sh_elem  = (threadIdx.x & 1) * 16;    // element offset 0 or 16

    // async global->LDS stage of one 64x32 B tile into buffer `buf`
    auto stage = [&](int kb, int buf) {
        int n = n0 + sn_local;
        if (n >= N) n = N - 1;                      // clamp tail (stores are guarded)
        const bf16_t* src = W + (size_t)n * K + kb + sh_elem;
        const unsigned lds_off =
            (unsigned)(uintptr_t)(&ldsB[buf][sn_local * 4 + (sh_elem >> 3)]);
        const unsigned long long gaddr = (unsigned long long)(uintptr_t)src;
        asm volatile(
            "global_load_async_to_lds_b128 %0, %1, off\n\t"
            "global_load_async_to_lds_b128 %0, %1, off offset:16"
            :: "v"(lds_off), "v"(gaddr) : "memory");
    };

    v8f acc[4] = {};

    stage(0, 0);
    asm volatile("s_wait_asynccnt 0x0" ::: "memory");
    __syncthreads();

    int cur = 0;
    for (int kb = 0; kb < K; kb += 32) {
        // prefetch next B tile into the other buffer (overlaps WMMAs below)
        if (kb + 32 < K) stage(kb + 32, cur ^ 1);

        // ---- A fragment (16x32 bf16, ISA 7.12.2 layout) ----
        // lane<16 : row m0+lane,    K = kb+{0..7, 16..23}
        // lane>=16: row m0+lane-16, K = kb+{8..15, 24..31}
        union { v16bf v; uint4 q[2]; } afrag;
        {
            const int row = m0 + (lane & 15);
            const int kof = kb + ((lane >> 4) << 3);
            const bf16_t* ap = A + (size_t)row * K + kof;
            afrag.q[0] = *(const uint4*)(ap);
            afrag.q[1] = *(const uint4*)(ap + 16);
            __builtin_prefetch((const void*)(ap + 32), 0, 3);
        }

        // ---- 4 B fragments from LDS + WMMA ----
        // B frag (32x16): lane n = lane&15, K = kb + (lane>=16 ? 8 : 0) + {0..7, 16..23}
        const uint4* bb = ldsB[cur];
        const int nrow_base = (lane & 15);
        const int kq        = (lane >> 4);          // uint4 offset: 0 or 1
#pragma unroll
        for (int nt = 0; nt < 4; ++nt) {
            union { v16bf v; uint4 q[2]; } bfrag;
            const int n_local = nt * 16 + nrow_base;
            bfrag.q[0] = bb[n_local * 4 + kq + 0];
            bfrag.q[1] = bb[n_local * 4 + kq + 2];
            acc[nt] = __builtin_amdgcn_wmma_f32_16x16x32_bf16(
                false, afrag.v, false, bfrag.v, (short)0, acc[nt], false, false);
        }

        // my async stores landed; barrier covers everyone's stage + tile reads
        asm volatile("s_wait_asynccnt 0x0" ::: "memory");
        __syncthreads();
        cur ^= 1;
    }

    // ---- epilogue: C/D layout => lane<16 rows m0+0..7, lane>=16 rows m0+8..15 ----
    const int colBase = n0 + (lane & 15);
    const int rowBase = m0 + ((lane >> 4) << 3);
#pragma unroll
    for (int nt = 0; nt < 4; ++nt) {
        const int n = colBase + nt * 16;
        if (n >= N) continue;
#pragma unroll
        for (int r = 0; r < 8; ++r) {
            const int m = rowBase + r;
            const float c = acc[nt][r];
            if constexpr (EPI == 0) {
                Cf[(size_t)m * N + n] = c;
            } else if constexpr (EPI == 1) {
                Cb0[(size_t)m * N + n] = __float2bfloat16(c);
            } else if constexpr (EPI == 2) {
                if (n < splitN) Cb0[(size_t)m * splitN + n] = __float2bfloat16(c);
                else            Cb1[(size_t)m * splitN + (n - splitN)] = __float2bfloat16(c);
            } else { // EPI == 3: softplus(c + bias)
                const float t = c + bias[n];
                const float sp = (t > 20.f) ? t : log1pf(__expf(t));
                Cf[(size_t)m * N + n] = sp;
            }
        }
    }
}

// ---------------- depthwise causal conv (k=4) + bias + SiLU ----------------
__global__ __launch_bounds__(256) void conv_silu_kernel(const bf16_t* __restrict__ x,
                                                        const float* __restrict__ cw,
                                                        const float* __restrict__ cb,
                                                        bf16_t* __restrict__ xs)
{
    size_t idx = (size_t)blockIdx.x * 256 + threadIdx.x;
    if (idx >= (size_t)MTOT * DI) return;
    const int d   = (int)(idx & (DI - 1));
    const size_t bl = idx >> 10;                // b*SEQ + l
    const int l   = (int)(bl & (SEQ - 1));
    const size_t bbase = bl - (size_t)l;        // b*SEQ

    float s = cb[d];
#pragma unroll
    for (int j = 0; j < 4; ++j) {
        const int ll = l - 3 + j;
        if (ll >= 0)
            s += cw[(size_t)d * 4 + j] *
                 __bfloat162float(x[(bbase + (size_t)ll) * DI + d]);
    }
    const float sig = 1.f / (1.f + __expf(-s));
    xs[idx] = __float2bfloat16(s * sig);
}

// ---------------- extract dt-rank columns of x_dbl as bf16 ----------------
__global__ __launch_bounds__(256) void extract_dtlow(const float* __restrict__ xdbl,
                                                     bf16_t* __restrict__ dtlow)
{
    size_t i = (size_t)blockIdx.x * 256 + threadIdx.x;
    if (i >= (size_t)MTOT * DTR) return;
    const size_t m = i >> 5;
    const int    r = (int)(i & (DTR - 1));
    dtlow[i] = __float2bfloat16(xdbl[m * NXPROJ + r]);
}

// ---------------- selective scan ----------------
// 256 blocks x 256 threads. Block handles (b, 32 channels). 8 lanes/channel,
// each lane owns 8 contiguous states; C-contraction reduced via shfl_xor.
__global__ __launch_bounds__(256)
void scan_kernel(const float* __restrict__ dt, const float* __restrict__ xdbl,
                 const bf16_t* __restrict__ xs, const float* __restrict__ A_log,
                 float* __restrict__ yraw)
{
    const int b    = blockIdx.x >> 5;
    const int d0   = (blockIdx.x & 31) * 32;
    const int wid  = threadIdx.x >> 5;
    const int lane = threadIdx.x & 31;
    const int ch   = d0 + wid * 4 + (lane >> 3);
    const int s0   = (lane & 7) * 8;

    float Aneg[8], h[8];
#pragma unroll
    for (int i = 0; i < 8; ++i) {
        Aneg[i] = -__expf(A_log[(size_t)ch * DSTATE + s0 + i]);
        h[i]    = 0.f;
    }

    const size_t rowBL = (size_t)b * SEQ;
    for (int l = 0; l < SEQ; ++l) {
        const size_t m  = rowBL + l;
        const float dtv = dt[m * DI + ch];
        const float u   = __bfloat162float(xs[m * DI + ch]);
        const float du  = dtv * u;

        const float4* Bp = (const float4*)(xdbl + m * NXPROJ + DTR + s0);
        const float4* Cp = (const float4*)(xdbl + m * NXPROJ + DTR + DSTATE + s0);
        const float4 B0 = Bp[0], B1 = Bp[1], C0 = Cp[0], C1 = Cp[1];
        const float Bt[8] = {B0.x,B0.y,B0.z,B0.w,B1.x,B1.y,B1.z,B1.w};
        const float Ct[8] = {C0.x,C0.y,C0.z,C0.w,C1.x,C1.y,C1.z,C1.w};

        float acc = 0.f;
#pragma unroll
        for (int i = 0; i < 8; ++i) {
            h[i] = h[i] * __expf(dtv * Aneg[i]) + du * Bt[i];
            acc += h[i] * Ct[i];
        }
        acc += __shfl_xor(acc, 1, 32);
        acc += __shfl_xor(acc, 2, 32);
        acc += __shfl_xor(acc, 4, 32);
        if ((lane & 7) == 0) yraw[m * DI + ch] = acc;
    }
}

// ---------------- y = (yraw + xs*D) * silu(z) -> bf16 ----------------
__global__ __launch_bounds__(256) void combine_kernel(const float* __restrict__ yraw,
                                                      const bf16_t* __restrict__ xs,
                                                      const bf16_t* __restrict__ z,
                                                      const float* __restrict__ Dv,
                                                      bf16_t* __restrict__ ybf)
{
    size_t i = (size_t)blockIdx.x * 256 + threadIdx.x;
    if (i >= (size_t)MTOT * DI) return;
    const int d = (int)(i & (DI - 1));
    float v = yraw[i] + __bfloat162float(xs[i]) * Dv[d];
    const float zf = __bfloat162float(z[i]);
    v *= zf / (1.f + __expf(-zf));
    ybf[i] = __float2bfloat16(v);
}

// ---------------- mean & max pool over L ----------------
__global__ __launch_bounds__(256) void pool_kernel(const float* __restrict__ enc,
                                                   float* __restrict__ out)
{
    const int i = blockIdx.x * 256 + threadIdx.x;
    if (i >= BATCH * DM) return;
    const int b = i / DM, d = i % DM;
    const float* p = enc + (size_t)b * SEQ * DM + d;
    float s = 0.f, mx = -3.402823466e38f;
    for (int l = 0; l < SEQ; ++l) {
        const float v = p[(size_t)l * DM];
        s += v;
        mx = fmaxf(mx, v);
    }
    out[(size_t)b * 2 * DM + d]      = s * (1.f / SEQ);
    out[(size_t)b * 2 * DM + DM + d] = mx;
}

// ---------------- host launcher ----------------
extern "C" void kernel_launch(void* const* d_in, const int* in_sizes, int n_in,
                              void* d_out, int out_size, void* d_ws, size_t ws_size,
                              hipStream_t stream)
{
    (void)in_sizes; (void)n_in; (void)out_size; (void)ws_size;

    const float* reads  = (const float*)d_in[0];
    const float* W_in   = (const float*)d_in[1];
    const float* conv_w = (const float*)d_in[2];
    const float* conv_b = (const float*)d_in[3];
    const float* W_xp   = (const float*)d_in[4];
    const float* W_dt   = (const float*)d_in[5];
    const float* b_dt   = (const float*)d_in[6];
    const float* A_log  = (const float*)d_in[7];
    const float* Dvec   = (const float*)d_in[8];
    const float* W_out  = (const float*)d_in[9];
    float* out = (float*)d_out;

    char* ws = (char*)d_ws;
    size_t off = 0;
    auto carve = [&](size_t bytes) -> char* {
        char* p = ws + off;
        off += (bytes + 255) & ~(size_t)255;
        return p;
    };

    bf16_t* rd_bf    = (bf16_t*)carve((size_t)MTOT * DM * 2);
    bf16_t* win_bf   = (bf16_t*)carve((size_t)2 * DI * DM * 2);
    bf16_t* wxp_bf   = (bf16_t*)carve((size_t)NXPROJ * DI * 2);
    bf16_t* wdt_bf   = (bf16_t*)carve((size_t)DI * DTR * 2);
    bf16_t* wout_bf  = (bf16_t*)carve((size_t)DM * DI * 2);
    bf16_t* x_bf     = (bf16_t*)carve((size_t)MTOT * DI * 2);
    bf16_t* z_bf     = (bf16_t*)carve((size_t)MTOT * DI * 2);
    bf16_t* xs_bf    = (bf16_t*)carve((size_t)MTOT * DI * 2);
    float*  xdbl_f   = (float*) carve((size_t)MTOT * NXPROJ * 4);
    bf16_t* dtlow_bf = (bf16_t*)carve((size_t)MTOT * DTR * 2);
    float*  dt_f     = (float*) carve((size_t)MTOT * DI * 4);
    float*  yraw_f   = (float*) carve((size_t)MTOT * DI * 4);
    bf16_t* y_bf     = (bf16_t*)carve((size_t)MTOT * DI * 2);
    float*  enc_f    = (float*) carve((size_t)MTOT * DM * 4);

    auto cvt = [&](const float* src, bf16_t* dst, size_t n) {
        cvt_f32_bf16<<<(unsigned)((n + 255) / 256), 256, 0, stream>>>(src, dst, n);
    };

    // 0) convert activations + weights to bf16
    cvt(reads, rd_bf,   (size_t)MTOT * DM);
    cvt(W_in,  win_bf,  (size_t)2 * DI * DM);
    cvt(W_xp,  wxp_bf,  (size_t)NXPROJ * DI);
    cvt(W_dt,  wdt_bf,  (size_t)DI * DTR);
    cvt(W_out, wout_bf, (size_t)DM * DI);

    const int mTiles = MTOT / 64;   // 256

    // 1) xz = reads @ W_in^T ; split into x / z (bf16)
    gemm_bf16_wmma<2><<<dim3(mTiles, (2 * DI) / 64), 128, 0, stream>>>(
        rd_bf, win_bf, MTOT, 2 * DI, DM, nullptr, x_bf, z_bf, nullptr, DI);

    // 2) depthwise causal conv + bias + SiLU
    conv_silu_kernel<<<(unsigned)(((size_t)MTOT * DI + 255) / 256), 256, 0, stream>>>(
        x_bf, conv_w, conv_b, xs_bf);

    // 3) x_dbl = xs @ W_xproj^T (fp32, N=160 -> 3 N-tiles with tail guard)
    gemm_bf16_wmma<0><<<dim3(mTiles, (NXPROJ + 63) / 64), 128, 0, stream>>>(
        xs_bf, wxp_bf, MTOT, NXPROJ, DI, xdbl_f, nullptr, nullptr, nullptr, 0);

    // 4) dt = softplus(x_dbl[:, :32] @ W_dt^T + b_dt)
    extract_dtlow<<<(unsigned)(((size_t)MTOT * DTR + 255) / 256), 256, 0, stream>>>(
        xdbl_f, dtlow_bf);
    gemm_bf16_wmma<3><<<dim3(mTiles, DI / 64), 128, 0, stream>>>(
        dtlow_bf, wdt_bf, MTOT, DI, DTR, dt_f, nullptr, nullptr, b_dt, 0);

    // 5) selective scan
    scan_kernel<<<BATCH * (DI / 32), 256, 0, stream>>>(dt_f, xdbl_f, xs_bf, A_log, yraw_f);

    // 6) gate: y = (yraw + xs*D) * silu(z)
    combine_kernel<<<(unsigned)(((size_t)MTOT * DI + 255) / 256), 256, 0, stream>>>(
        yraw_f, xs_bf, z_bf, Dvec, y_bf);

    // 7) enc = y @ W_out^T (fp32)
    gemm_bf16_wmma<0><<<dim3(mTiles, DM / 64), 128, 0, stream>>>(
        y_bf, wout_bf, MTOT, DM, DI, enc_f, nullptr, nullptr, nullptr, 0);

    // 8) mean/max pool over L -> (B, 2*DM)
    pool_kernel<<<(BATCH * DM + 255) / 256, 256, 0, stream>>>(enc_f, out);
}
